// ClimateAwareDeformableAligner_10222022165176
// MI455X (gfx1250) — matrix operand order repaired and checked by reading, not compiled, not run on hardware
//
#include <hip/hip_runtime.h>
#include <hip/hip_bf16.h>
#include <stdint.h>

// ---------------------------------------------------------------------------
// ClimateAwareDeformableAligner for MI455X (gfx1250), wave32 + WMMA + async-LDS.
//
// Ref math collapses: grid_sample has W==1 and offsets feed the W coord, so
//   out1[b,l,c] = x[b,l,c] + tanh(gate)*0.5*(x[b,2047,c] + x[b,2048,c])
// offset output = tanh(conv3(gelu(conv2(gelu(GN(conv1(pin)))))))*10
// Convs run as WMMA f16-in/f32-acc GEMMs (v_wmma_f32_16x16x32_f16); global->LDS
// staging uses GLOBAL_LOAD_ASYNC_TO_LDS_B128 (ASYNCcnt) for the bulk channels.
// ---------------------------------------------------------------------------

#define Bdim 16
#define Ldim 4096
#define Cdim 512
#define Ndim 128   // B*G

typedef __attribute__((ext_vector_type(16))) _Float16 v16h;
typedef __attribute__((ext_vector_type(8)))  float    v8f;

__device__ __forceinline__ v8f wmma16(v16h a, v16h b, v8f c) {
  // 8 args: (neg_a, A, neg_b, B, c_mod, C, reuse_a, reuse_b)
  return __builtin_amdgcn_wmma_f32_16x16x32_f16(false, a, false, b, (short)0, c,
                                                false, false);
}

// gfx1250 async global->LDS copy, 16 bytes per lane. LDS dest offset is the
// low 32 bits of the generic pointer (LDS aperture occupies addr[63:32]).
__device__ __forceinline__ void async_load_b128(void* lds_dst,
                                                const float* src) {
  unsigned loff = (unsigned)(uintptr_t)lds_dst;
  unsigned long long ga = (unsigned long long)(uintptr_t)src;
  asm volatile("global_load_async_to_lds_b128 %0, %1, off" ::"v"(loff), "v"(ga)
               : "memory");
}
__device__ __forceinline__ void wait_async0() {
  asm volatile("s_wait_asynccnt 0" ::: "memory");
}

// Build B fragment (32x16, f16) from LDS tile staged as [pos][KCH] f16 with
// KCH even and zero-padded channels. K index = tap*KCH + ch (tap-major).
// B k order used here (contiguous, same as build_A): V_j, lanes 0-15 hold K=2j,2j+1 (col=lane), lanes 16-31
// hold K=16+2j,16+2j+1. Pairs (K,K+1) = consecutive channels of one tap ->
// single aligned 32-bit LDS read.
template <int KCH, int KTOT>
__device__ __forceinline__ v16h build_B(const _Float16* lds, int pos_base,
                                        int chunk) {
  const int lane = threadIdx.x & 31;
  const int col = lane & 15;
  const int kbase = chunk * 32 + ((lane & 16) ? 16 : 0);
  union { v16h v; unsigned u[8]; } b;
  const unsigned* l32 = (const unsigned*)lds;
#pragma unroll
  for (int j = 0; j < 8; ++j) {
    const int K = kbase + 2 * j;
    unsigned val = 0u;
    if (K < KTOT) {
      const int tap = K / KCH;        // KCH compile-time
      const int ch = K - tap * KCH;   // even
      const int pos = pos_base + col + tap;
      val = l32[(pos * KCH + ch) >> 1];
    }
    b.u[j] = val;
  }
  return b.v;
}

// Build A fragment (16x32, f16) from fp32 conv weights w[OC][IC][3].
// A k order used here (contiguous, same as build_B): row m = lane&15; half h -> VGPR j=h>>1;
//   K = (lane>=16 ? 16 : 0) + h.
template <int IC, int KCH, int KTOT>
__device__ __forceinline__ v16h build_A(const float* __restrict__ w, int mbase,
                                        int chunk) {
  const int lane = threadIdx.x & 31;
  const int m = lane & 15;
  const int hi = (lane & 16) ? 16 : 0;
  union { v16h v; _Float16 e[16]; } a;
#pragma unroll
  for (int h = 0; h < 16; ++h) {
    const int j = h >> 1;
    const int K = chunk * 32 + hi + h;
    float f = 0.f;
    if (K < KTOT) {
      const int tap = K / KCH;
      const int ch = K - tap * KCH;
      if (ch < IC) f = w[((mbase + m) * IC + ch) * 3 + tap];
    }
    a.e[h] = (_Float16)f;
  }
  return a.v;
}

__device__ __forceinline__ float gelu_exact(float x) {
  return 0.5f * x * (1.f + erff(x * 0.70710678118654752f));
}

// ---------------------------------------------------------------------------
// Stage pin = concat(xg, x_ext-tile) for positions [p_first, p_first+NP):
//  1) async-copy the 64 x-channels (f32) into lds_stage[NP][64]
//  2) convert to packed f16 pin[NP][72] (ch 68..71 pad = 0, OOB rows = 0),
//     folding in the 4 x_ext channels.
// Caller must __syncthreads() after return before reading lds_pin.
// ---------------------------------------------------------------------------
template <int NP>
__device__ __forceinline__ void stage_pin_async(
    _Float16* lds_pin, float* lds_stage, int p_first, int n,
    const float* __restrict__ x, const float* __restrict__ x_ext) {
  const int b = n >> 3;   // n / G
  const int g = n & 7;    // n % G
  const int b2 = n & 15;  // n % B (reference's tile() indexing)
  for (int idx = threadIdx.x; idx < NP * 16; idx += 128) {
    const int p = idx >> 4, c4 = idx & 15;
    const int l = p_first + p;
    if ((unsigned)l < (unsigned)Ldim) {
      async_load_b128(lds_stage + p * 64 + c4 * 4,
                      x + ((size_t)(b * Ldim + l) * Cdim + g * 64 + c4 * 4));
    }
  }
  wait_async0();
  __syncthreads();  // async data from all waves visible
  for (int idx = threadIdx.x; idx < NP * 36; idx += 128) {
    const int p = idx / 36, cp = idx - p * 36;
    const int ch = cp * 2;
    const int l = p_first + p;
    float v0 = 0.f, v1 = 0.f;
    if ((unsigned)l < (unsigned)Ldim) {
      if (ch < 64) {
        const float2 vv = *(const float2*)(lds_stage + p * 64 + ch);
        v0 = vv.x; v1 = vv.y;
      } else if (ch < 68) {
        v0 = x_ext[(b2 * Ldim + l) * 4 + (ch - 64)];
        v1 = x_ext[(b2 * Ldim + l) * 4 + (ch - 63)];
      }
    }
    union { _Float16 h[2]; unsigned u; } pk;
    pk.h[0] = (_Float16)v0;
    pk.h[1] = (_Float16)v1;
    ((unsigned*)lds_pin)[p * 36 + cp] = pk.u;
  }
}

// ---------------------------------------------------------------------------
// Kernel 0: scalar prep (tanh of gate), avoids 8.4M redundant tanhf's.
// ---------------------------------------------------------------------------
__global__ void prep_kernel(const float* __restrict__ gate,
                            float* __restrict__ tg) {
  if (threadIdx.x == 0) tg[0] = tanhf(gate[0]);
}

// ---------------------------------------------------------------------------
// Kernel 1: main output. Streaming: out = x + tg*0.5*(row2047+row2048).
// ---------------------------------------------------------------------------
__global__ void __launch_bounds__(256) main_out_kernel(
    const float* __restrict__ x, const float* __restrict__ tgp,
    float* __restrict__ out) {
  const int i = blockIdx.x * 256 + threadIdx.x;  // float4 index
  const int total = Bdim * Ldim * (Cdim / 4);
  if (i >= total) return;
  const float tg = tgp[0];
  const int c4 = i & (Cdim / 4 - 1);
  const int b = i >> 19;  // / (4096*128)
  const float4* x4 = (const float4*)x;
  const float4 v  = x4[i];
  const float4 r0 = x4[(b * Ldim + 2047) * (Cdim / 4) + c4];
  const float4 r1 = x4[(b * Ldim + 2048) * (Cdim / 4) + c4];
  float4 o;
  o.x = v.x + tg * 0.5f * (r0.x + r1.x);
  o.y = v.y + tg * 0.5f * (r0.y + r1.y);
  o.z = v.z + tg * 0.5f * (r0.z + r1.z);
  o.w = v.w + tg * 0.5f * (r0.w + r1.w);
  ((float4*)out)[i] = o;
}

// ---------------------------------------------------------------------------
// Kernel 2: conv1 (WMMA) -> per-(n,group) partial sum/sumsq for GroupNorm.
// Grid: 128 n * 16 blocks of 256 positions (4 staged sub-segments of 64);
// 4 waves, wave w owns out-channel tile w == GN group w. A-fragments built
// once per block and reused across 16 l-tiles.
// ---------------------------------------------------------------------------
__global__ void __launch_bounds__(128) conv1_stats_kernel(
    const float* __restrict__ x, const float* __restrict__ x_ext,
    const float* __restrict__ w1, const float* __restrict__ b1,
    float* __restrict__ partials) {
  __shared__ __align__(16) unsigned char smem[9504 + 16896];
  _Float16* lds_pin = (_Float16*)smem;            // [66][72] f16
  float* lds_stage = (float*)(smem + 9504);       // [66][64] f32

  const int blk = blockIdx.x;
  const int n = blk >> 4, bs = blk & 15;
  const int tid = threadIdx.x, w = tid >> 5, lane = tid & 31;

  const int mbase = w * 16;
  v16h A[7];
#pragma unroll
  for (int c = 0; c < 7; ++c) A[c] = build_A<68, 72, 216>(w1, mbase, c);
  const int hi = (lane & 16) ? 8 : 0;
  float bias[8];
#pragma unroll
  for (int v = 0; v < 8; ++v) bias[v] = b1[mbase + hi + v];

  float s = 0.f, q = 0.f;
  for (int sub = 0; sub < 4; ++sub) {
    const int s0 = bs * 256 + sub * 64;
    stage_pin_async<66>(lds_pin, lds_stage, s0 - 1, n, x, x_ext);
    __syncthreads();
    for (int lt = 0; lt < 4; ++lt) {
      v8f cacc = {};
#pragma unroll
      for (int c = 0; c < 7; ++c)
        cacc = wmma16(A[c], build_B<72, 216>(lds_pin, lt * 16, c), cacc);
#pragma unroll
      for (int v = 0; v < 8; ++v) {
        const float val = cacc[v] + bias[v];
        s += val;
        q += val * val;
      }
    }
    __syncthreads();  // before next sub overwrites pin/stage
  }
  // deterministic wave32 tree reduction
#pragma unroll
  for (int off = 16; off; off >>= 1) {
    s += __shfl_xor(s, off, 32);
    q += __shfl_xor(q, off, 32);
  }
  if (lane == 0) {
    const int pi = ((n * 4 + w) * 16 + bs) * 2;
    partials[pi] = s;
    partials[pi + 1] = q;
  }
}

// ---------------------------------------------------------------------------
// Kernel 3: finalize GroupNorm stats: mean, rstd per (n, group). 512 entries.
// ---------------------------------------------------------------------------
__global__ void stats_finalize_kernel(const float* __restrict__ partials,
                                      float* __restrict__ stats) {
  const int t = threadIdx.x;
  if (t >= Ndim * 4) return;
  float s = 0.f, q = 0.f;
  for (int i = 0; i < 16; ++i) {
    s += partials[(t * 16 + i) * 2];
    q += partials[(t * 16 + i) * 2 + 1];
  }
  const float inv = 1.f / (16.f * 4096.f);
  const float mean = s * inv;
  const float var = q * inv - mean * mean;
  stats[t * 2] = mean;
  stats[t * 2 + 1] = rsqrtf(var + 1e-5f);
}

// ---------------------------------------------------------------------------
// Kernel 4: fused conv1(WMMA)+GN+GELU -> conv2(WMMA)+GELU -> conv3+tanh*10.
// Block = (n, 64-position segment). LDS layout (39.2 KB): the f32 async
// landing zone is dead after conversion and aliases the h/h2 regions.
// h and h2 zeroed outside [0,L) to model conv1d zero padding between stages.
// ---------------------------------------------------------------------------
__global__ void __launch_bounds__(128) deform_offset_kernel(
    const float* __restrict__ x, const float* __restrict__ x_ext,
    const float* __restrict__ w1, const float* __restrict__ b1,
    const float* __restrict__ gn_g, const float* __restrict__ gn_b,
    const float* __restrict__ w2, const float* __restrict__ b2,
    const float* __restrict__ w3, const float* __restrict__ stats,
    float* __restrict__ offset_out) {
  __shared__ __align__(16) unsigned char smem[14112 + 25088];
  _Float16* lds_pin  = (_Float16*)smem;                    // [98][72] f16
  float*    lds_stage = (float*)(smem + 14112);            // [98][64] f32 (temp)
  _Float16* lds_h    = (_Float16*)(smem + 14112);          // [96][64] f16
  float*    lds_h2   = (float*)(smem + 14112 + 12288);     // [80][32] f32

  const int blk = blockIdx.x;
  const int n = blk >> 6, seg = blk & 63;
  const int tid = threadIdx.x, w = tid >> 5, lane = tid & 31;
  const int s0 = seg * 64;
  const int hi = (lane & 16) ? 8 : 0;
  const int col = lane & 15;

  stage_pin_async<98>(lds_pin, lds_stage, s0 - 17, n, x, x_ext);
  __syncthreads();  // stage region now dead; pin ready

  // ---- phase 1: conv1 + GroupNorm + GELU -> lds_h (wave w = ch-tile/group w)
  {
    const int mbase = w * 16;
    v16h A[7];
#pragma unroll
    for (int c = 0; c < 7; ++c) A[c] = build_A<68, 72, 216>(w1, mbase, c);
    float bias[8], gam[8], bet[8];
#pragma unroll
    for (int v = 0; v < 8; ++v) {
      const int row = mbase + hi + v;
      bias[v] = b1[row]; gam[v] = gn_g[row]; bet[v] = gn_b[row];
    }
    const float mean = stats[(n * 4 + w) * 2];
    const float rstd = stats[(n * 4 + w) * 2 + 1];
    for (int lt = 0; lt < 6; ++lt) {
      v8f cacc = {};
#pragma unroll
      for (int c = 0; c < 7; ++c)
        cacc = wmma16(A[c], build_B<72, 216>(lds_pin, lt * 16, c), cacc);
      const int hp = lt * 16 + col;
      const int l = s0 - 16 + hp;
      const bool inr = (unsigned)l < (unsigned)Ldim;
#pragma unroll
      for (int v = 0; v < 8; ++v) {
        float val = (cacc[v] + bias[v] - mean) * rstd * gam[v] + bet[v];
        val = gelu_exact(val);
        lds_h[hp * 64 + (mbase + hi + v)] = (_Float16)(inr ? val : 0.f);
      }
    }
  }
  __syncthreads();

  // ---- phase 2: conv2 + GELU -> lds_h2 (2 M-tiles x 5 l-tiles over 4 waves)
  {
    const int mt = w & 1;
    const int mbase = mt * 16;
    v16h A2[6];
#pragma unroll
    for (int c = 0; c < 6; ++c) A2[c] = build_A<64, 64, 192>(w2, mbase, c);
    float bias[8];
#pragma unroll
    for (int v = 0; v < 8; ++v) bias[v] = b2[mbase + hi + v];
    for (int lt = (w >> 1); lt < 5; lt += 2) {
      v8f cacc = {};
#pragma unroll
      for (int c = 0; c < 6; ++c)
        cacc = wmma16(A2[c], build_B<64, 192>(lds_h, lt * 16 + 7, c), cacc);
      const int p2 = lt * 16 + col;
      const int l2 = s0 - 8 + p2;
      const bool inr = (unsigned)l2 < (unsigned)Ldim;
#pragma unroll
      for (int v = 0; v < 8; ++v) {
        const float val = gelu_exact(cacc[v] + bias[v]);
        lds_h2[p2 * 32 + (mbase + hi + v)] = inr ? val : 0.f;
      }
    }
  }
  __syncthreads();

  // ---- phase 3: conv3 (32->1, k=3) + tanh*ORF (tiny; VALU)
  if (tid < 64) {
    const int l = s0 + tid;
    float acc = 0.f;
#pragma unroll
    for (int tap = 0; tap < 3; ++tap) {
      const int p2 = tid + 7 + tap;
#pragma unroll
      for (int ch = 0; ch < 32; ++ch)
        acc += w3[ch * 3 + tap] * lds_h2[p2 * 32 + ch];
    }
    offset_out[n * Ldim + l] = tanhf(acc) * 10.f;
  }
}

// ---------------------------------------------------------------------------
extern "C" void kernel_launch(void* const* d_in, const int* in_sizes, int n_in,
                              void* d_out, int out_size, void* d_ws,
                              size_t ws_size, hipStream_t stream) {
  const float* x     = (const float*)d_in[0];
  const float* x_ext = (const float*)d_in[1];
  const float* w1    = (const float*)d_in[2];
  const float* b1    = (const float*)d_in[3];
  const float* gn_g  = (const float*)d_in[4];
  const float* gn_b  = (const float*)d_in[5];
  const float* w2    = (const float*)d_in[6];
  const float* b2    = (const float*)d_in[7];
  const float* w3    = (const float*)d_in[8];
  const float* gate  = (const float*)d_in[9];

  float* out1    = (float*)d_out;
  float* out_off = out1 + (size_t)Bdim * Ldim * Cdim;  // offset (128,1,4096)

  float* partials = (float*)d_ws;              // 128*4*16*2 floats
  float* statsb   = partials + Ndim * 4 * 16 * 2;
  float* tg       = statsb + Ndim * 4 * 2;

  prep_kernel<<<1, 32, 0, stream>>>(gate, tg);
  main_out_kernel<<<(Bdim * Ldim * Cdim / 4 + 255) / 256, 256, 0, stream>>>(
      x, tg, out1);
  conv1_stats_kernel<<<Ndim * 16, 128, 0, stream>>>(x, x_ext, w1, b1, partials);
  stats_finalize_kernel<<<1, 512, 0, stream>>>(partials, statsb);
  deform_offset_kernel<<<Ndim * 64, 128, 0, stream>>>(
      x, x_ext, w1, b1, gn_g, gn_b, w2, b2, w3, statsb, out_off);
}
